// LinearAttention_11905649344761
// MI455X (gfx1250) — matrix operand order repaired and checked by reading, not compiled, not run on hardware
//
#include <hip/hip_runtime.h>

typedef __attribute__((ext_vector_type(16))) _Float16 v16h;
typedef __attribute__((ext_vector_type(8)))  _Float16 v8h;
typedef __attribute__((ext_vector_type(8)))  float    v8f;

union V16 { v16h v; v8h h[2]; _Float16 e[16]; };

// Problem sizes from the reference setup_inputs()
constexpr int N_ = 8, L_ = 4096, H_ = 16, E_ = 64;
constexpr int HEADS = N_ * H_;                    // 128
constexpr float TEMP_ = 0.35355339059327379f;     // 64^-0.25
constexpr float EPS_  = 1e-6f;

constexpr int SPLITS = 8;                         // sequence splits in phase 1
constexpr int CHUNK  = 32;                        // K-dim of one WMMA
constexpr int S_PER_SPLIT = L_ / SPLITS;          // 512
constexpr int CHUNKS_PER_SPLIT = S_PER_SPLIT / CHUNK; // 16

__device__ __forceinline__ float feat(float x) {
  // elu(x * temp) + 1 == (t>0 ? t+1 : exp(t)) with t = x*temp
  float t = x * TEMP_;
  return t > 0.0f ? t + 1.0f : __expf(t);
}

__global__ void zero_ws_kernel(float* __restrict__ ws, int n) {
  int i = blockIdx.x * blockDim.x + threadIdx.x;
  if (i < n) ws[i] = 0.0f;
}

// ---------------------------------------------------------------------------
// Phase 1: KV[head][m][d] = sum_s phi(k[s,d]) * v[s,m]    (64x64 per head)
//          ksum[head][d]  = sum_s phi(k[s,d])
// grid = HEADS*SPLITS blocks, 512 threads (16 waves); wave w owns tile
// (mi = w/4, di = w%4) of the 64x64 output; accumulation via f32 atomics.
// ---------------------------------------------------------------------------
__global__ __launch_bounds__(512) void lin_attn_kv_kernel(
    const float* __restrict__ K, const float* __restrict__ V,
    float* __restrict__ kv, float* __restrict__ ksum)
{
  __shared__ alignas(16) _Float16 ldsKT[64][CHUNK];   // [d][s], phi(k)
  __shared__ alignas(16) _Float16 ldsVT[64][CHUNK];   // [m][s], v

  const int head  = blockIdx.x / SPLITS;
  const int split = blockIdx.x % SPLITS;
  const int n = head / H_, h = head % H_;

  const int t    = threadIdx.x;
  const int lane = t & 31;
  const int wave = t >> 5;                  // 0..15
  const int mi   = wave >> 2;               // 0..3 (m tile)
  const int di   = wave & 3;                // 0..3 (d tile)

  // cooperative loader mapping: 512 threads cover 32 rows x 64 cols (x4 floats)
  const int ls = t >> 4;                    // s within chunk, 0..31
  const int lc = (t & 15) * 4;              // starting d/m column, 0..60

  const size_t rowStride = (size_t)H_ * E_; // stride between sequence rows
  const size_t base = ((size_t)n * L_ * H_ + (size_t)h) * E_;

  v8f acc = {};                             // this wave's 16x16 f32 tile
  float sk0 = 0.f, sk1 = 0.f, sk2 = 0.f, sk3 = 0.f;

  const int s0 = split * S_PER_SPLIT;

  const int mrow = mi * 16 + (lane & 15);
  const int dcol = di * 16 + (lane & 15);
  const int koff = (lane < 16) ? 0 : 8;     // A-matrix 16-bit layout
  const int hk   = (lane < 16) ? 0 : 16;    // unused; B loads use koff, same order as A

  for (int c = 0; c < CHUNKS_PER_SPLIT; ++c) {
    const int sbase = s0 + c * CHUNK;
    const float* kp = K + base + (size_t)(sbase + ls) * rowStride + lc;
    const float* vp = V + base + (size_t)(sbase + ls) * rowStride + lc;
    float4 kq = *(const float4*)kp;
    float4 vq = *(const float4*)vp;
    // prefetch next chunk (speculative; dropped harmlessly past the end)
    __builtin_prefetch(kp + CHUNK * rowStride, 0, 1);
    __builtin_prefetch(vp + CHUNK * rowStride, 0, 1);

    __syncthreads();   // previous iteration's LDS reads complete

    float f0 = feat(kq.x), f1 = feat(kq.y), f2 = feat(kq.z), f3 = feat(kq.w);
    sk0 += f0; sk1 += f1; sk2 += f2; sk3 += f3;
    ldsKT[lc + 0][ls] = (_Float16)f0;
    ldsKT[lc + 1][ls] = (_Float16)f1;
    ldsKT[lc + 2][ls] = (_Float16)f2;
    ldsKT[lc + 3][ls] = (_Float16)f3;
    ldsVT[lc + 0][ls] = (_Float16)vq.x;
    ldsVT[lc + 1][ls] = (_Float16)vq.y;
    ldsVT[lc + 2][ls] = (_Float16)vq.z;
    ldsVT[lc + 3][ls] = (_Float16)vq.w;

    __syncthreads();

    // A = V^T tile (16x32): lane row = mrow, halves per 16-bit A layout
    V16 a, b;
    a.h[0] = *(const v8h*)&ldsVT[mrow][koff];
    a.h[1] = *(const v8h*)&ldsVT[mrow][koff + 16];
    // B = phi(K) tile (32x16): lane col = dcol, s halves matching A layout
    b.h[0] = *(const v8h*)&ldsKT[dcol][koff];
    b.h[1] = *(const v8h*)&ldsKT[dcol][koff + 16];

    acc = __builtin_amdgcn_wmma_f32_16x16x32_f16(
        false, a.v, false, b.v, (short)0, acc, false, false);
  }

  // accumulate this split's partial KV tile into workspace
  float* kvh = kv + (size_t)head * 64 * 64;
  const int nn = lane & 15;
  const int rbase = (lane < 16) ? 0 : 8;
#pragma unroll
  for (int r = 0; r < 8; ++r) {
    int m = mi * 16 + rbase + r;
    atomicAdd(&kvh[m * 64 + di * 16 + nn], acc[r]);
  }
  // ksum partials (4 columns per thread)
  float* ksh = ksum + (size_t)head * 64;
  atomicAdd(&ksh[lc + 0], sk0);
  atomicAdd(&ksh[lc + 1], sk1);
  atomicAdd(&ksh[lc + 2], sk2);
  atomicAdd(&ksh[lc + 3], sk3);
}

// ---------------------------------------------------------------------------
// Phase 2: out[n,h,l,m] = z[l] * sum_d phi(q[l,d]) * KV[m,d]
//          z[l] = 1 / (sum_d phi(q[l,d]) * ksum[d] + eps)
// grid = HEADS*32 blocks, 256 threads (8 waves); each wave = one 16-row tile.
// ---------------------------------------------------------------------------
__global__ __launch_bounds__(256) void lin_attn_out_kernel(
    const float* __restrict__ Q, const float* __restrict__ kv,
    const float* __restrict__ ksum, float* __restrict__ out)
{
  __shared__ alignas(16) _Float16 ldsKV[64][64];  // [m][d] as f16
  __shared__ float ldsKS[64];

  const int head = blockIdx.x >> 5;   // 32 blocks per head
  const int blk  = blockIdx.x & 31;
  const int n = head / H_, h = head % H_;
  const int t = threadIdx.x, lane = t & 31, wave = t >> 5;

  // stage this head's KV (f32 -> f16) and ksum into LDS
  const float* kvh = kv + (size_t)head * 64 * 64;
#pragma unroll
  for (int i = 0; i < 16; ++i) {
    int idx = t + i * 256;
    ldsKV[idx >> 6][idx & 63] = (_Float16)kvh[idx];
  }
  if (t < 64) ldsKS[t] = ksum[(size_t)head * 64 + t];
  __syncthreads();

  const int lbase = blk * 128 + wave * 16;       // first l row of this tile
  const int l = lbase + (lane & 15);
  const float* qrow = Q + ((size_t)n * L_ + l) * H_ * E_ + (size_t)h * E_;

  const int koff = (lane < 16) ? 0 : 8;

  // Build A (two 16x32 chunks over d) and the normalizer dot in one pass
  V16 a[2];
  float p = 0.0f;
#pragma unroll
  for (int ch = 0; ch < 2; ++ch) {
    const int dA = ch * 32 + koff;
#pragma unroll
    for (int j = 0; j < 8; ++j) {
      float f0 = feat(qrow[dA + j]);
      float f1 = feat(qrow[dA + 16 + j]);
      p += f0 * ldsKS[dA + j] + f1 * ldsKS[dA + 16 + j];
      a[ch].e[j]     = (_Float16)f0;
      a[ch].e[8 + j] = (_Float16)f1;
    }
  }
  // lanes L and L^16 hold complementary halves of row (lane&15)'s d values
  float pfull = p + __shfl_xor(p, 16, 32);
  float z = 1.0f / (pfull + EPS_);               // valid in lane (row id)

  const int hk    = (lane < 16) ? 0 : 16;
  const int nn    = lane & 15;
  const int rbase = (lane < 16) ? 0 : 8;
  float zr[8];
#pragma unroll
  for (int r = 0; r < 8; ++r) zr[r] = __shfl(z, rbase + r, 32);

  float* orow = out + (size_t)(n * H_ + h) * L_ * 64;   // out[n][h][l][m]

#pragma unroll
  for (int mt = 0; mt < 4; ++mt) {
    v8f accum = {};
#pragma unroll
    for (int ch = 0; ch < 2; ++ch) {
      V16 b;   // B = KV^T (32x16): lane col m = mt*16+nn, d halves matching A layout
      b.h[0] = *(const v8h*)&ldsKV[mt * 16 + nn][ch * 32 + koff];
      b.h[1] = *(const v8h*)&ldsKV[mt * 16 + nn][ch * 32 + koff + 16];
      accum = __builtin_amdgcn_wmma_f32_16x16x32_f16(
          false, a[ch].v, false, b.v, (short)0, accum, false, false);
    }
#pragma unroll
    for (int r = 0; r < 8; ++r) {
      int row = lbase + rbase + r;
      orow[(size_t)row * 64 + mt * 16 + nn] = accum[r] * zr[r];
    }
  }
}

extern "C" void kernel_launch(void* const* d_in, const int* in_sizes, int n_in,
                              void* d_out, int out_size, void* d_ws, size_t ws_size,
                              hipStream_t stream) {
  const float* Q = (const float*)d_in[0];
  const float* K = (const float*)d_in[1];
  const float* V = (const float*)d_in[2];
  float* out = (float*)d_out;

  float* ws   = (float*)d_ws;
  float* kv   = ws;                               // HEADS * 64 * 64
  float* ksum = ws + (size_t)HEADS * 64 * 64;     // HEADS * 64
  const int ws_elems = HEADS * 64 * 64 + HEADS * 64;

  zero_ws_kernel<<<(ws_elems + 255) / 256, 256, 0, stream>>>(ws, ws_elems);

  lin_attn_kv_kernel<<<HEADS * SPLITS, 512, 0, stream>>>(K, V, kv, ksum);

  lin_attn_out_kernel<<<HEADS * 32, 256, 0, stream>>>(Q, kv, ksum, out);

  (void)in_sizes; (void)n_in; (void)out_size; (void)ws_size;
}
